// AC_88399016886986
// MI455X (gfx1250) — matrix operand order repaired and checked by reading, not compiled, not run on hardware
//
#include <hip/hip_runtime.h>

// ---------------- CDNA5 types ----------------
typedef __attribute__((ext_vector_type(16))) __bf16 v16bf;
typedef __attribute__((ext_vector_type(8)))  float  v8f;

// ---------------- problem constants ----------------
#define NNODES  16384
#define NEDGES  262144
#define DMODEL  256
#define DFFN    1024
#define EDIMC   160
#define NGRAPH  64
#define NLAYER  3
#define SCALE_QK 0.17677669529663687f   // 1/sqrt(32)

// ---------------- bf16 helpers (round-to-nearest-even) ----------------
__device__ __forceinline__ __bf16 f2bf(float x) {
    unsigned u = __float_as_uint(x);
    u += 0x7FFFu + ((u >> 16) & 1u);
    unsigned short h = (unsigned short)(u >> 16);
    return __builtin_bit_cast(__bf16, h);
}
__device__ __forceinline__ float bf2f(__bf16 b) {
    unsigned short h = __builtin_bit_cast(unsigned short, b);
    return __uint_as_float(((unsigned)h) << 16);
}

union Frag { uint4 q[2]; v16bf v; };

// async global->LDS copy of 16B (CDNA5; tracked by ASYNCcnt)
__device__ __forceinline__ void async_copy_b128(unsigned lds_addr, const void* gaddr) {
    asm volatile("global_load_async_to_lds_b128 %0, %1, off"
                 :: "v"(lds_addr), "v"((unsigned long long)(size_t)gaddr)
                 : "memory");
}
__device__ __forceinline__ void wait_asynccnt0() {
    asm volatile("s_wait_asynccnt 0x0" ::: "memory");
}

// ==========================================================================
// Weight packing: W[Nout,K] row-major f32 -> bf16 WMMA B-fragment layout.
// Tile (16 N x 32 K) = 512 elems laid out [lane(32)][elem(16)]:
//   n = 16*ntile + (lane&15),  k = 32*ktile + (lane>>4)*16 + elem
// Tile id = ntile * (K/32) + ktile.
// ==========================================================================
__global__ __launch_bounds__(256) void ac_pack(const float* __restrict__ W,
                                               __bf16* __restrict__ dst,
                                               int Nout, int K) {
    int idx = blockIdx.x * 256 + threadIdx.x;
    int tile = idx >> 9;
    int r    = idx & 511;
    int l    = r >> 4;
    int e    = r & 15;
    int kt   = K >> 5;
    int n = (tile / kt) * 16 + (l & 15);
    int k = (tile % kt) * 32 + ((l >> 4) << 4) + e;
    dst[idx] = f2bf(W[(size_t)n * K + k]);
}

// ==========================================================================
// Input prep
// ==========================================================================
__global__ __launch_bounds__(256) void ac_cat(const float* __restrict__ pw,
                                              const float* __restrict__ bm,
                                              __bf16* __restrict__ out) {
    int t = blockIdx.x * 256 + threadIdx.x;           // over N*160
    int n = t / EDIMC, r = t % EDIMC;
    int rb = r >> 4, j = r & 15;
    float v = (j < 8) ? pw[(size_t)(n * 10 + rb) * 8 + j]
                      : bm[(size_t)(n * 10 + rb) * 8 + (j - 8)];
    out[t] = f2bf(v);
}

__global__ __launch_bounds__(256) void ac_cvt(const float* __restrict__ in,
                                              __bf16* __restrict__ out, int n) {
    int t = blockIdx.x * 256 + threadIdx.x;
    if (t < n) out[t] = f2bf(in[t]);
}

// ==========================================================================
// bf16 WMMA GEMM:  C[M,Nout] = A[M,K] (bf16, row-major) x Bpacked^T + bias
// Block: 256 thr = 8 waves. Block tile 128M x 64N, K-step 32.
// A tiles double-buffered in LDS via GLOBAL_LOAD_ASYNC_TO_LDS_B128:
//   one barrier per K-step; async issue for tile k+1 happens right after the
//   barrier (prior DS reads of that buffer completed before the barrier since
//   __syncthreads lowers with s_wait_dscnt 0 before s_barrier_signal).
// ==========================================================================
__global__ __launch_bounds__(256) void ac_gemm(const __bf16* __restrict__ A,
                                               const __bf16* __restrict__ Bp,
                                               const float* __restrict__ bias,
                                               float* __restrict__ Cf,
                                               __bf16* __restrict__ Cbf,
                                               int M, int Nout, int K, int relu) {
    __shared__ __align__(16) __bf16 As[2][128 * 32];
    const int tid  = threadIdx.x;
    const int wave = tid >> 5, lane = tid & 31;
    const int wm = wave & 3, wn = wave >> 2;
    const int m0 = blockIdx.x * 128;
    const int n0 = blockIdx.y * 64;
    const int ktiles = K >> 5;

    // this thread's two 16B chunks of the 128x32 tile
    const int c0 = tid,        row0 = c0 >> 2, cc0 = c0 & 3;
    const int c1 = tid + 256,  row1 = c1 >> 2, cc1 = c1 & 3;
    const __bf16* Abase = A + (size_t)m0 * K;

    auto issue = [&](int kt, int buf) {
        const __bf16* g = Abase + (size_t)kt * 32;
        unsigned l0 = (unsigned)(size_t)&As[buf][row0 * 32 + cc0 * 8];
        unsigned l1 = (unsigned)(size_t)&As[buf][row1 * 32 + cc1 * 8];
        async_copy_b128(l0, g + (size_t)row0 * K + cc0 * 8);
        async_copy_b128(l1, g + (size_t)row1 * K + cc1 * 8);
    };

    v8f acc[2][2] = {};
    const int half = lane >> 4, mloc = lane & 15;

    issue(0, 0);                       // prologue: async-fill buffer 0
    for (int kt = 0; kt < ktiles; ++kt) {
        const int buf = kt & 1;
        wait_asynccnt0();              // own async copies for tile kt done
        __syncthreads();               // all waves' copies visible in LDS
        if (kt + 1 < ktiles) issue(kt + 1, buf ^ 1);   // prefetch next tile

        // A fragments from LDS (contiguous k = 16*half + elem, matching packed B)
        Frag fa[2];
#pragma unroll
        for (int i = 0; i < 2; ++i) {
            const __bf16* ar = &As[buf][(wm * 32 + i * 16 + mloc) * 32];
            fa[i].q[0] = *(const uint4*)(ar + half * 16);
            fa[i].q[1] = *(const uint4*)(ar + half * 16 + 8);
        }
        // B fragments straight from L2-resident packed weights
        Frag fb[2];
#pragma unroll
        for (int j = 0; j < 2; ++j) {
            int ntile = (n0 + wn * 32 + j * 16) >> 4;
            const __bf16* bp = Bp + ((size_t)(ntile * ktiles + kt) << 9) + lane * 16;
            fb[j].q[0] = *(const uint4*)(bp);
            fb[j].q[1] = *(const uint4*)(bp + 8);
        }
#pragma unroll
        for (int i = 0; i < 2; ++i)
#pragma unroll
            for (int j = 0; j < 2; ++j)
                acc[i][j] = __builtin_amdgcn_wmma_f32_16x16x32_bf16(
                    false, fa[i].v, false, fb[j].v, (short)0, acc[i][j], false, false);
    }

    // epilogue: C/D layout -> m = i*16 + r + (lane>=16)*8, n = lane&15
    const int nloc = lane & 15;
#pragma unroll
    for (int i = 0; i < 2; ++i)
#pragma unroll
        for (int j = 0; j < 2; ++j) {
            int n = n0 + wn * 32 + j * 16 + nloc;
            float bi = bias ? bias[n] : 0.f;
#pragma unroll
            for (int r = 0; r < 8; ++r) {
                int m = m0 + wm * 32 + i * 16 + half * 8 + r;
                float v = acc[i][j][r] + bi;
                if (relu) v = fmaxf(v, 0.f);
                if (Cf)  Cf[(size_t)m * Nout + n] = v;
                if (Cbf) Cbf[(size_t)m * Nout + n] = f2bf(v);
            }
        }
}

// ==========================================================================
// Elementwise / attention kernels (wave32)
// ==========================================================================
__global__ __launch_bounds__(256) void ac_addin(const float* __restrict__ x,
                                                const float* __restrict__ inp0,
                                                float* __restrict__ xin,
                                                __bf16* __restrict__ xinbf) {
    int t = blockIdx.x * 256 + threadIdx.x;           // N*256
    float v = x[t] + inp0[t];
    xin[t] = v; xinbf[t] = f2bf(v);
}

__global__ __launch_bounds__(256) void ac_init(float* __restrict__ mbuf,
                                               float* __restrict__ den,
                                               float* __restrict__ agg) {
    int t = blockIdx.x * 256 + threadIdx.x;           // N*256
    agg[t] = 0.f;
    if (t < NNODES * 8) { mbuf[t] = -__builtin_inff(); den[t] = 0.f; }
}

// alpha[e,h] = SCALE * sum_d q[dst][h,d] * (k[src][h,d] + ee[e][h,d])
__global__ __launch_bounds__(256) void ac_alpha(const float* __restrict__ qn,
                                                const float* __restrict__ kn,
                                                const __bf16* __restrict__ ee,
                                                const int* __restrict__ src,
                                                const int* __restrict__ dst,
                                                float* __restrict__ alpha) {
    int e = blockIdx.x * 8 + (threadIdx.x >> 5);
    int lane = threadIdx.x & 31;
    if (e >= NEDGES) return;
    int s = src[e], d = dst[e];
    const float*  qp = qn + (size_t)d * DMODEL;
    const float*  kp = kn + (size_t)s * DMODEL;
    const __bf16* ep = ee + (size_t)e * DMODEL;
#pragma unroll
    for (int h = 0; h < 8; ++h) {
        int c = h * 32 + lane;
        float v = qp[c] * (kp[c] + bf2f(ep[c]));
#pragma unroll
        for (int off = 16; off; off >>= 1) v += __shfl_xor(v, off);
        if (lane == 0) alpha[(size_t)e * 8 + h] = v * SCALE_QK;
    }
}

__device__ __forceinline__ void atomMaxF(float* p, float v) {
    if (v >= 0.f) atomicMax((int*)p, __float_as_int(v));
    else          atomicMin((unsigned int*)p, __float_as_uint(v));
}

__global__ __launch_bounds__(256) void ac_segmax(const float* __restrict__ alpha,
                                                 const int* __restrict__ dst,
                                                 float* __restrict__ mbuf) {
    int t = blockIdx.x * 256 + threadIdx.x;           // E*8
    int e = t >> 3, h = t & 7;
    atomMaxF(&mbuf[(size_t)dst[e] * 8 + h], alpha[t]);
}

__global__ __launch_bounds__(256) void ac_expden(const float* __restrict__ alpha,
                                                 const int* __restrict__ dst,
                                                 const float* __restrict__ mbuf,
                                                 float* __restrict__ den,
                                                 float* __restrict__ ex) {
    int t = blockIdx.x * 256 + threadIdx.x;           // E*8
    int e = t >> 3, h = t & 7;
    int d = dst[e];
    float a = __expf(alpha[t] - mbuf[(size_t)d * 8 + h]);
    ex[t] = a;
    atomicAdd(&den[(size_t)d * 8 + h], a);
}

// agg[dst] += a * (v[src] + ee[e]); one block per edge, thread per channel
__global__ __launch_bounds__(256) void ac_agg(const float* __restrict__ ex,
                                              const float* __restrict__ den,
                                              const float* __restrict__ vn,
                                              const __bf16* __restrict__ ee,
                                              const int* __restrict__ src,
                                              const int* __restrict__ dst,
                                              float* __restrict__ agg) {
    int e = blockIdx.x, c = threadIdx.x;
    int s = src[e], d = dst[e];
    int h = c >> 5;
    float a = ex[(size_t)e * 8 + h] / (den[(size_t)d * 8 + h] + 1e-16f);
    float v = a * (vn[(size_t)s * DMODEL + c] + bf2f(ee[(size_t)e * DMODEL + c]));
    atomicAdd(&agg[(size_t)d * DMODEL + c], v);
}

// LayerNorm over D=256: y = LN(A + B [+ C]) * g + b ; one wave per row
__global__ __launch_bounds__(256) void ac_ln(const float* __restrict__ Av,
                                             const float* __restrict__ Bv,
                                             const float* __restrict__ Cv,
                                             const float* __restrict__ g,
                                             const float* __restrict__ b,
                                             float* __restrict__ Yf,
                                             __bf16* __restrict__ Ybf) {
    int row = blockIdx.x * 8 + (threadIdx.x >> 5);
    int lane = threadIdx.x & 31;
    size_t base = (size_t)row * DMODEL;
    float vals[8], s = 0.f;
#pragma unroll
    for (int i = 0; i < 8; ++i) {
        int c = i * 32 + lane;
        float v = Av[base + c] + Bv[base + c];
        if (Cv) v += Cv[base + c];
        vals[i] = v; s += v;
    }
#pragma unroll
    for (int off = 16; off; off >>= 1) s += __shfl_xor(s, off);
    float mean = s * (1.f / 256.f);
    float vs = 0.f;
#pragma unroll
    for (int i = 0; i < 8; ++i) { float d = vals[i] - mean; vs += d * d; }
#pragma unroll
    for (int off = 16; off; off >>= 1) vs += __shfl_xor(vs, off);
    float inv = rsqrtf(vs * (1.f / 256.f) + 1e-5f);
#pragma unroll
    for (int i = 0; i < 8; ++i) {
        int c = i * 32 + lane;
        float y = (vals[i] - mean) * inv * g[c] + b[c];
        Yf[base + c] = y;
        if (Ybf) Ybf[base + c] = f2bf(y);
    }
}

// heads: copy x -> h_padded, link logits, mask (all nodes valid: dense batches)
__global__ __launch_bounds__(256) void ac_link(const float* __restrict__ x,
                                               const float* __restrict__ wl,
                                               const float* __restrict__ bl,
                                               float* __restrict__ outh,
                                               float* __restrict__ logits,
                                               float* __restrict__ maskout) {
    int row = blockIdx.x * 8 + (threadIdx.x >> 5);
    int lane = threadIdx.x & 31;
    const float* xr = x + (size_t)row * DMODEL;
    float s = 0.f;
#pragma unroll
    for (int i = 0; i < 8; ++i) {
        int c = i * 32 + lane;
        float v = xr[c];
        outh[(size_t)row * DMODEL + c] = v;
        s += v * wl[c];
    }
#pragma unroll
    for (int off = 16; off; off >>= 1) s += __shfl_xor(s, off);
    if (lane == 0) { logits[row] = s + bl[0]; maskout[row] = 1.0f; }
}

__global__ __launch_bounds__(256) void ac_values(const float* __restrict__ x,
                                                 const float* __restrict__ wc,
                                                 const float* __restrict__ bc,
                                                 float* __restrict__ values) {
    __shared__ float red[256];
    int g = blockIdx.x, d = threadIdx.x;
    const float* xg = x + (size_t)g * 256 * DMODEL;
    float s = 0.f;
    for (int n = 0; n < 256; ++n) s += xg[(size_t)n * DMODEL + d];
    red[d] = s * wc[d];
    __syncthreads();
    for (int off = 128; off; off >>= 1) {
        if (d < off) red[d] += red[d + off];
        __syncthreads();
    }
    if (d == 0) values[g] = red[0] * (1.f / 256.f) + bc[0];
}

// ==========================================================================
// host driver
// ==========================================================================
extern "C" void kernel_launch(void* const* d_in, const int* in_sizes, int n_in,
                              void* d_out, int out_size, void* d_ws, size_t ws_size,
                              hipStream_t stream) {
    const float* power = (const float*)d_in[0];
    const float* beam  = (const float*)d_in[1];
    const float* npa   = (const float*)d_in[2];
    const float* epa   = (const float*)d_in[3];
    const float* W_in  = (const float*)d_in[4];
    const float* b_in  = (const float*)d_in[5];
    const float* W_emb = (const float*)d_in[6];
    const float* b_emb = (const float*)d_in[7];
    const float* Wq = (const float*)d_in[8];   const float* bq = (const float*)d_in[9];
    const float* Wk = (const float*)d_in[10];  const float* bk = (const float*)d_in[11];
    const float* Wv = (const float*)d_in[12];  const float* bv = (const float*)d_in[13];
    const float* We = (const float*)d_in[14];
    const float* Wsk = (const float*)d_in[15]; const float* bsk = (const float*)d_in[16];
    const float* ln1g = (const float*)d_in[17]; const float* ln1b = (const float*)d_in[18];
    const float* Wf1 = (const float*)d_in[19]; const float* bf1 = (const float*)d_in[20];
    const float* Wf2 = (const float*)d_in[21]; const float* bf2 = (const float*)d_in[22];
    const float* ln2g = (const float*)d_in[23]; const float* ln2b = (const float*)d_in[24];
    const float* W_link = (const float*)d_in[25]; const float* b_link = (const float*)d_in[26];
    const float* W_crit = (const float*)d_in[27]; const float* b_crit = (const float*)d_in[28];
    const int* ei = (const int*)d_in[29];
    const int* srcp = ei;
    const int* dstp = ei + NEDGES;
    (void)in_sizes; (void)n_in; (void)out_size; (void)ws_size;

    // ---- workspace carve-up ----
    char* ws = (char*)d_ws;
    size_t off = 0;
    auto alloc = [&](size_t bytes) -> void* {
        void* p = ws + off;
        off = (off + bytes + 255) & ~(size_t)255;
        return p;
    };

    // packed-weight region offsets (elements)
    const size_t OFF_WIN  = 0;
    const size_t OFF_WEMB = OFF_WIN + 256 * 160;
    const size_t OFF_L0   = OFF_WEMB + 256 * 160;
    const size_t LQ = 0, LK = 65536, LV = 131072, LS = 196608,
                 LE = 262144, LF1 = 303104, LF2 = 565248;
    const size_t LAYER_W = 827392;
    const size_t WP_ELEMS = OFF_L0 + NLAYER * LAYER_W;

    __bf16* wp      = (__bf16*)alloc(WP_ELEMS * 2);
    __bf16* in0cat  = (__bf16*)alloc((size_t)NNODES * EDIMC * 2);
    __bf16* npabf   = (__bf16*)alloc((size_t)NNODES * EDIMC * 2);
    __bf16* epabf   = (__bf16*)alloc((size_t)NEDGES * EDIMC * 2);
    float*  inp0    = (float*)alloc((size_t)NNODES * DMODEL * 4);
    float*  xbuf    = (float*)alloc((size_t)NNODES * DMODEL * 4);
    float*  xin     = (float*)alloc((size_t)NNODES * DMODEL * 4);
    __bf16* xinbf   = (__bf16*)alloc((size_t)NNODES * DMODEL * 2);
    float*  qn      = (float*)alloc((size_t)NNODES * DMODEL * 4);
    float*  kn      = (float*)alloc((size_t)NNODES * DMODEL * 4);
    float*  vn      = (float*)alloc((size_t)NNODES * DMODEL * 4);
    float*  xsk     = (float*)alloc((size_t)NNODES * DMODEL * 4);
    __bf16* eebuf   = (__bf16*)alloc((size_t)NEDGES * DMODEL * 2);
    float*  alpha   = (float*)alloc((size_t)NEDGES * 8 * 4);
    float*  exbuf   = (float*)alloc((size_t)NEDGES * 8 * 4);
    float*  mbuf    = (float*)alloc((size_t)NNODES * 8 * 4);
    float*  den     = (float*)alloc((size_t)NNODES * 8 * 4);
    float*  agg     = (float*)alloc((size_t)NNODES * DMODEL * 4);
    float*  h1      = (float*)alloc((size_t)NNODES * DMODEL * 4);
    __bf16* h1bf    = (__bf16*)alloc((size_t)NNODES * DMODEL * 2);
    __bf16* f1bf    = (__bf16*)alloc((size_t)NNODES * DFFN * 2);
    float*  f2v     = (float*)alloc((size_t)NNODES * DMODEL * 4);

    float* out      = (float*)d_out;
    float* out_h    = out;                                  // [N,256]
    float* out_lg   = out + (size_t)NNODES * DMODEL;        // [N]
    float* out_val  = out_lg + NNODES;                      // [B]
    float* out_mask = out_val + NGRAPH;                     // [N]

    auto packW = [&](const float* Wsrc, size_t dstOff, int Nout, int K) {
        ac_pack<<<(Nout * K) / 256, 256, 0, stream>>>(Wsrc, wp + dstOff, Nout, K);
    };
    auto gemm = [&](const __bf16* A, size_t bOff, const float* bias,
                    float* Cf, __bf16* Cbf, int M, int Nout, int K, int relu) {
        dim3 grid(M / 128, Nout / 64);
        ac_gemm<<<grid, 256, 0, stream>>>(A, wp + bOff, bias, Cf, Cbf, M, Nout, K, relu);
    };

    // ---- pack all weights to WMMA B-fragment layout (stay hot in L2) ----
    packW(W_in,  OFF_WIN,  256, 160);
    packW(W_emb, OFF_WEMB, 256, 160);
    for (int l = 0; l < NLAYER; ++l) {
        size_t LB = OFF_L0 + (size_t)l * LAYER_W;
        packW(Wq  + (size_t)l * 256 * 256,  LB + LQ,  256, 256);
        packW(Wk  + (size_t)l * 256 * 256,  LB + LK,  256, 256);
        packW(Wv  + (size_t)l * 256 * 256,  LB + LV,  256, 256);
        packW(Wsk + (size_t)l * 256 * 256,  LB + LS,  256, 256);
        packW(We  + (size_t)l * 256 * 160,  LB + LE,  256, 160);
        packW(Wf1 + (size_t)l * 1024 * 256, LB + LF1, 1024, 256);
        packW(Wf2 + (size_t)l * 256 * 1024, LB + LF2, 256, 1024);
    }

    // ---- activation prep (fp32 -> bf16 stream tensors) ----
    ac_cat<<<(NNODES * EDIMC) / 256, 256, 0, stream>>>(power, beam, in0cat);
    ac_cvt<<<(NNODES * EDIMC) / 256, 256, 0, stream>>>(npa, npabf, NNODES * EDIMC);
    ac_cvt<<<(NEDGES * EDIMC) / 256, 256, 0, stream>>>(epa, epabf, NEDGES * EDIMC);

    // ---- input linears ----
    gemm(in0cat, OFF_WIN,  b_in,  inp0, nullptr, NNODES, 256, 160, 0);
    gemm(npabf,  OFF_WEMB, b_emb, xbuf, nullptr, NNODES, 256, 160, 0);

    // ---- transformer layers ----
    for (int l = 0; l < NLAYER; ++l) {
        size_t LB = OFF_L0 + (size_t)l * LAYER_W;
        ac_addin<<<(NNODES * DMODEL) / 256, 256, 0, stream>>>(xbuf, inp0, xin, xinbf);

        gemm(xinbf, LB + LQ, bq + l * 256,  qn,  nullptr, NNODES, 256, 256, 0);
        gemm(xinbf, LB + LK, bk + l * 256,  kn,  nullptr, NNODES, 256, 256, 0);
        gemm(xinbf, LB + LV, bv + l * 256,  vn,  nullptr, NNODES, 256, 256, 0);
        gemm(xinbf, LB + LS, bsk + l * 256, xsk, nullptr, NNODES, 256, 256, 0);
        gemm(epabf, LB + LE, nullptr, nullptr, eebuf, NEDGES, 256, 160, 0);

        ac_init<<<NNODES, 256, 0, stream>>>(mbuf, den, agg);
        ac_alpha<<<NEDGES / 8, 256, 0, stream>>>(qn, kn, eebuf, srcp, dstp, alpha);
        ac_segmax<<<(NEDGES * 8) / 256, 256, 0, stream>>>(alpha, dstp, mbuf);
        ac_expden<<<(NEDGES * 8) / 256, 256, 0, stream>>>(alpha, dstp, mbuf, den, exbuf);
        ac_agg<<<NEDGES, 256, 0, stream>>>(exbuf, den, vn, eebuf, srcp, dstp, agg);

        ac_ln<<<NNODES / 8, 256, 0, stream>>>(xin, agg, xsk,
                                              ln1g + l * 256, ln1b + l * 256, h1, h1bf);
        gemm(h1bf, LB + LF1, bf1 + l * 1024, nullptr, f1bf, NNODES, 1024, 256, 1);
        gemm(f1bf, LB + LF2, bf2 + l * 256,  f2v, nullptr, NNODES, 256, 1024, 0);
        ac_ln<<<NNODES / 8, 256, 0, stream>>>(h1, f2v, nullptr,
                                              ln2g + l * 256, ln2b + l * 256, xbuf, nullptr);
    }

    // ---- heads ----
    ac_link<<<NNODES / 8, 256, 0, stream>>>(xbuf, W_link, b_link, out_h, out_lg, out_mask);
    ac_values<<<NGRAPH, 256, 0, stream>>>(xbuf, W_crit, b_crit, out_val);
}
